// CustomLSTM_10033043604131
// MI455X (gfx1250) — compile-verified, matrix operand order checked by reading, not run on hardware
//
#include <hip/hip_runtime.h>

typedef __attribute__((ext_vector_type(16))) __bf16 v16bf;
typedef __attribute__((ext_vector_type(8)))  __bf16 v8bf;
typedef __attribute__((ext_vector_type(4)))  __bf16 v4bf;
typedef __attribute__((ext_vector_type(2)))  __bf16 v2bf;
typedef __attribute__((ext_vector_type(8)))  float  v8f;
typedef __attribute__((ext_vector_type(4)))  float  v4f;

#define B_ROWS 32768
#define D_INF  1024
#define H_DIM  512
#define BM 128
#define BN 64
#define BK 32
#define KP 40            // padded K stride (elements): 80B row, multiple of 16B
#define NK (D_INF / BK)  // 32 K-steps
#define NTHREADS 512     // 16 waves

__device__ __forceinline__ float fast_sigmoid(float x) {
    return 1.0f / (1.0f + __expf(-x));
}
__device__ __forceinline__ float fast_tanh(float x) {
    return 2.0f / (1.0f + __expf(-2.0f * x)) - 1.0f;
}

__launch_bounds__(NTHREADS)
__global__ void lstm_gates_wmma_bf16(const float* __restrict__ x,
                                     const float* __restrict__ cprev,
                                     const float* __restrict__ Wi, const float* __restrict__ bi,
                                     const float* __restrict__ Wf, const float* __restrict__ bf,
                                     const float* __restrict__ Wg, const float* __restrict__ bg,
                                     const float* __restrict__ Wo, const float* __restrict__ bo,
                                     float* __restrict__ out_h,
                                     float* __restrict__ out_c)
{
    __shared__ __bf16 lds_a[2][BM][KP];        // activations, [M][K], double-buffered
    __shared__ __bf16 lds_w[2][4][BN][KP];     // weights TRANSPOSED [gate][N][K], double-buffered

    const int tid    = threadIdx.x;
    const int lane   = tid & 31;
    const int wave   = tid >> 5;               // 0..15
    const int waveM  = wave >> 2;              // 0..3  (32 rows each)
    const int waveN  = wave & 3;               // 0..3  (16 cols each)
    const int laneLo = lane & 15;
    const int laneHi = lane >> 4;              // 0..1
    const int klo    = laneHi * 8;             // per-lane K chunk (ISA bf16 A/B layout)

    const int rowBase = blockIdx.y * BM;
    const int colBase = blockIdx.x * BN;

    // ---- W staging assignment: one gate per 128-thread group, 4(K)x4(N) micro-tile ----
    const int gate = tid >> 7;                 // 0..3, wave-uniform
    const int tt   = tid & 127;
    const int wkt  = (tt >> 4) * 4;            // K micro-tile base: 0..28
    const int wn4  = (tt & 15) * 4;            // N micro-tile base: 0..60
    const float* Wp = (gate == 0) ? Wi : (gate == 1) ? Wf : (gate == 2) ? Wg : Wo;

    v8f acc[4][2];
#pragma unroll
    for (int g = 0; g < 4; ++g)
#pragma unroll
        for (int mi = 0; mi < 2; ++mi)
            acc[g][mi] = {};

    // staging registers held across compute for latency hiding
    v4f aR[2];
    v4f wR[4];

    auto loadA = [&](int k0) {
#pragma unroll
        for (int i = 0; i < 2; ++i) {
            const int idx = tid + i * NTHREADS;       // BM*BK/4 = 1024 float4
            const int row = idx >> 3;
            const int c4  = idx & 7;
            aR[i] = *(const v4f*)(x + (size_t)(rowBase + row) * D_INF + k0 + c4 * 4);
        }
    };
    auto loadW = [&](int k0) {
#pragma unroll
        for (int j = 0; j < 4; ++j)
            wR[j] = *(const v4f*)(Wp + (size_t)(k0 + wkt + j) * H_DIM + colBase + wn4);
    };
    auto storeA = [&](int b) {
#pragma unroll
        for (int i = 0; i < 2; ++i) {
            const int idx = tid + i * NTHREADS;
            const int row = idx >> 3;
            const int c4  = idx & 7;
            v4bf p = { (__bf16)aR[i][0], (__bf16)aR[i][1],
                       (__bf16)aR[i][2], (__bf16)aR[i][3] };
            *(v4bf*)&lds_a[b][row][c4 * 4] = p;       // 8B-aligned ds_store_b64
        }
    };
    auto storeW = [&](int b) {
#pragma unroll
        for (int n = 0; n < 4; ++n)
#pragma unroll
            for (int kp = 0; kp < 2; ++kp) {
                v2bf p = { (__bf16)wR[2 * kp][n], (__bf16)wR[2 * kp + 1][n] };
                *(v2bf*)&lds_w[b][gate][wn4 + n][wkt + 2 * kp] = p;  // 4B-aligned ds_store_b32
            }
    };
    auto compute = [&](int b) {
        v16bf afrag[2];
#pragma unroll
        for (int mi = 0; mi < 2; ++mi) {
            const __bf16* ap = &lds_a[b][waveM * 32 + mi * 16 + laneLo][0];
            v8bf lo = *(const v8bf*)(ap + klo);          // K = klo..klo+7
            v8bf hi = *(const v8bf*)(ap + klo + 16);     // K = klo+16..klo+23
            afrag[mi] = __builtin_shufflevector(lo, hi,
                0,1,2,3,4,5,6,7,8,9,10,11,12,13,14,15);
        }
#pragma unroll
        for (int g = 0; g < 4; ++g) {
            const __bf16* bp = &lds_w[b][g][waveN * 16 + laneLo][0];
            v8bf lo = *(const v8bf*)(bp + klo);
            v8bf hi = *(const v8bf*)(bp + klo + 16);
            v16bf bfrag = __builtin_shufflevector(lo, hi,
                0,1,2,3,4,5,6,7,8,9,10,11,12,13,14,15);
#pragma unroll
            for (int mi = 0; mi < 2; ++mi)
                acc[g][mi] = __builtin_amdgcn_wmma_f32_16x16x32_bf16(
                    false, afrag[mi], false, bfrag, (short)0, acc[g][mi], false, false);
        }
    };

    // ---- software-pipelined main loop: 1 barrier per K-step, double-buffered LDS ----
    loadA(0);
    loadW(0);
    storeA(0);
    storeW(0);

    for (int k = 0; k < NK; ++k) {
        __syncthreads();
        const int buf = k & 1;
        if (k + 1 < NK) {
            loadA((k + 1) * BK);      // issue next-tile globals before compute
            loadW((k + 1) * BK);
            if (k + 2 < NK) {         // prefetch the K+2 activation tile (global_prefetch_b8)
                __builtin_prefetch(x + (size_t)(rowBase + (tid >> 2)) * D_INF
                                     + (k + 2) * BK + (tid & 3) * 8, 0, 0);
            }
        }
        compute(buf);                 // WMMA work hides global-load latency
        if (k + 1 < NK) {
            storeA(buf ^ 1);
            storeW(buf ^ 1);
        }
    }

    // ---- epilogue: bias + gate nonlinearities + fused cell/hidden update ----
    const int   col  = colBase + waveN * 16 + laneLo;   // fixed per lane (C layout: N = lane&15)
    const float bi_v = bi[col];
    const float bf_v = bf[col];
    const float bg_v = bg[col];
    const float bo_v = bo[col];

#pragma unroll
    for (int mi = 0; mi < 2; ++mi) {
#pragma unroll
        for (int r = 0; r < 8; ++r) {
            const int row = rowBase + waveM * 32 + mi * 16 + laneHi * 8 + r;
            const size_t off = (size_t)row * H_DIM + col;
            const float it = fast_sigmoid(acc[0][mi][r] + bi_v);
            const float ft = fast_sigmoid(acc[1][mi][r] + bf_v);
            const float gt = fast_tanh   (acc[2][mi][r] + bg_v);
            const float ot = fast_sigmoid(acc[3][mi][r] + bo_v);
            const float ct = ft * cprev[off] + it * gt;
            const float ht = ot * fast_tanh(ct);
            out_h[off] = ht;
            out_c[off] = ct;
        }
    }
}

extern "C" void kernel_launch(void* const* d_in, const int* in_sizes, int n_in,
                              void* d_out, int out_size, void* d_ws, size_t ws_size,
                              hipStream_t stream) {
    (void)in_sizes; (void)n_in; (void)out_size; (void)d_ws; (void)ws_size;

    const float* x     = (const float*)d_in[0];
    const float* cprev = (const float*)d_in[1];
    const float* Wi    = (const float*)d_in[2];
    const float* bi    = (const float*)d_in[3];
    const float* Wf    = (const float*)d_in[4];
    const float* bf    = (const float*)d_in[5];
    const float* Wg    = (const float*)d_in[6];
    const float* bg    = (const float*)d_in[7];
    const float* Wo    = (const float*)d_in[8];
    const float* bo    = (const float*)d_in[9];

    float* out_h = (float*)d_out;                                // h_t first
    float* out_c = (float*)d_out + (size_t)B_ROWS * H_DIM;       // then c_t

    dim3 grid(H_DIM / BN, B_ROWS / BM);   // (8, 256)
    dim3 block(NTHREADS);
    lstm_gates_wmma_bf16<<<grid, block, 0, stream>>>(
        x, cprev, Wi, bi, Wf, bf, Wg, bg, Wo, bo, out_h, out_c);
}
